// DilatedResidualBlock_54528904790900
// MI455X (gfx1250) — matrix operand order repaired and checked by reading, not compiled, not run on hardware
//
#include <hip/hip_runtime.h>
#include <hip/hip_bf16.h>

typedef _Float16 v16h __attribute__((ext_vector_type(16)));
typedef _Float16 v8h  __attribute__((ext_vector_type(8)));
typedef float    v8f  __attribute__((ext_vector_type(8)));

#define NPTS 16384L
#define NKF  262144L   // N*K
#define EPS_BN 1e-5f

__device__ __forceinline__ float lrelu_f(float v) { return v >= 0.f ? v : 0.01f * v; }

// ---------------------------------------------------------------------------
// Fold BN into conv weights (f32 -> f16), pad Cin if needed, fold bias (f32).
// ---------------------------------------------------------------------------
__global__ void fold_weights(const float* __restrict__ W, const float* __restrict__ b,
                             const float* __restrict__ g, const float* __restrict__ be,
                             _Float16* __restrict__ Wf, float* __restrict__ bf,
                             int Cout, int CinS, int CinD) {
    int i = blockIdx.x * blockDim.x + threadIdx.x;
    if (i >= Cout * CinD) return;
    int o = i / CinD, c = i % CinD;
    float s = g ? g[o] * rsqrtf(1.f + EPS_BN) : 1.f;
    Wf[i] = (c < CinS) ? (_Float16)(W[o * CinS + c] * s) : (_Float16)0.f;
    if (c == 0) bf[o] = b[o] * s + (be ? be[o] : 0.f);
}

__global__ void f32_to_f16(const float* __restrict__ src, _Float16* __restrict__ dst, long n) {
    long i = (long)blockIdx.x * blockDim.x + threadIdx.x;
    if (i < n) dst[i] = (_Float16)src[i];
}

// ---------------------------------------------------------------------------
// relpos = [dist, rel(3), center(3), neigh(3), zero-pad to 32] : f16 [32][NK]
// ---------------------------------------------------------------------------
__global__ void build_relpos(const float* __restrict__ xyz, const int* __restrict__ idx,
                             _Float16* __restrict__ rp) {
    long f = (long)blockIdx.x * blockDim.x + threadIdx.x;
    if (f >= NKF) return;
    long n = f >> 4;
    long j = (long)idx[f];
    float cx = xyz[n],            cy = xyz[NPTS + n],     cz = xyz[2 * NPTS + n];
    float nx = xyz[j],            ny = xyz[NPTS + j],     nz = xyz[2 * NPTS + j];
    float rx = cx - nx, ry = cy - ny, rz = cz - nz;
    float d = sqrtf(rx * rx + ry * ry + rz * rz);
    float v[10] = {d, rx, ry, rz, cx, cy, cz, nx, ny, nz};
#pragma unroll
    for (int r = 0; r < 10; ++r) rp[(long)r * NKF + f] = (_Float16)v[r];
#pragma unroll
    for (int r = 10; r < 32; ++r) rp[(long)r * NKF + f] = (_Float16)0.f;
}

// ---------------------------------------------------------------------------
// gather: dst[c][n*16+k] = src[c][idx[n*16+k]]   (f16, channel-major)
// ---------------------------------------------------------------------------
__global__ void gather_rows(const _Float16* __restrict__ src, const int* __restrict__ idx,
                            _Float16* __restrict__ dst, int C) {
    long f = (long)blockIdx.x * blockDim.x + threadIdx.x;
    if (f >= NKF) return;
    long j = (long)idx[f];
    for (int c = 0; c < C; ++c)
        dst[(long)c * NKF + f] = src[(long)c * NPTS + j];
}

// ---------------------------------------------------------------------------
// A-fragment loader: W row-major [Cout][CIN] f16 (ISA 16-bit A 16x32 layout).
// ---------------------------------------------------------------------------
__device__ __forceinline__ v16h load_afrag(const _Float16* wrow, int kb) {
    union { v16h v; v8h h[2]; } au;
    au.h[0] = *(const v8h*)(wrow + kb);
    au.h[1] = *(const v8h*)(wrow + 16 + kb);
    return au.v;
}

__device__ __forceinline__ void store_tile(void* Y, int mode, long Np, int ctile,
                                           long pt0, long pt1, int m_off, int cn,
                                           const v8f& acc0, const v8f& acc1) {
    if (mode == 0) {
        _Float16* Yh = (_Float16*)Y;
#pragma unroll
        for (int r = 0; r < 8; ++r) {
            long row = (long)(ctile * 16 + m_off + r) * Np;
            Yh[row + pt0 + cn] = (_Float16)lrelu_f(acc0[r]);
            Yh[row + pt1 + cn] = (_Float16)lrelu_f(acc1[r]);
        }
    } else if (mode == 1) {
        float* Yf = (float*)Y;
#pragma unroll
        for (int r = 0; r < 8; ++r) {
            long row = (long)(ctile * 16 + m_off + r) * Np;
            Yf[row + pt0 + cn] = acc0[r];
            Yf[row + pt1 + cn] = acc1[r];
        }
    } else {
        float* Yf = (float*)Y;
#pragma unroll
        for (int r = 0; r < 8; ++r) {
            long row = (long)(ctile * 16 + m_off + r) * Np;
            long o0 = row + pt0 + cn, o1 = row + pt1 + cn;
            Yf[o0] = lrelu_f(Yf[o0] + acc0[r]);
            Yf[o1] = lrelu_f(Yf[o1] + acc1[r]);
        }
    }
}

// ---------------------------------------------------------------------------
// 1x1-conv GEMM: Y[Cout][Np] = act(Wf[Cout][CIN] * X[CIN][Np] + bf)
// One wave = 2 point-tiles (32 points), dual accumulator chains share A frags.
// CIN <= 128: B fragments preloaded in registers across all Cout tiles.
// CIN == 256: stream B, but process 2 Cout tiles / iter (4 acc chains) so each
//             streamed B pair is used by two A fragments (halves X traffic).
// mode 0: f16+lrelu | 1: f32, no act | 2: f32 read-add-lrelu-store
// ---------------------------------------------------------------------------
template<int CIN>
__global__ void conv1x1_wmma(const _Float16* __restrict__ W, const float* __restrict__ bias,
                             const _Float16* __restrict__ X, void* __restrict__ Y,
                             int Cout, long Np, int mode) {
    constexpr int KT = CIN / 32;
    constexpr bool PRELOAD = (CIN <= 128);
    const int lane  = threadIdx.x & 31;
    const int wave  = threadIdx.x >> 5;
    const long pt0  = ((long)blockIdx.x * 8 + wave) * 32;
    const long pt1  = pt0 + 16;
    const int mrow  = lane & 15;
    const int m_off = (lane < 16) ? 0 : 8;
    const int kb    = (lane < 16) ? 0 : 8;
    const int cn    = lane & 15;

    if constexpr (PRELOAD) {
        v16h pb0[KT], pb1[KT];
#pragma unroll
        for (int kt = 0; kt < KT; ++kt) {
#pragma unroll
            for (int s = 0; s < 16; ++s) {
                const _Float16* xrow = X + (long)(kt * 32 + kb + (s & 7) + ((s >> 3) << 4)) * Np;
                pb0[kt][s] = xrow[pt0 + cn];
                pb1[kt][s] = xrow[pt1 + cn];
            }
        }
        for (int ct = 0; ct < (Cout >> 4); ++ct) {
            v8f acc0, acc1;
#pragma unroll
            for (int r = 0; r < 8; ++r) { float bv = bias[ct * 16 + m_off + r]; acc0[r] = bv; acc1[r] = bv; }
#pragma unroll
            for (int kt = 0; kt < KT; ++kt) {
                v16h a = load_afrag(W + (long)(ct * 16 + mrow) * CIN + kt * 32, kb);
                acc0 = __builtin_amdgcn_wmma_f32_16x16x32_f16(false, a, false, pb0[kt], (short)0, acc0, false, false);
                acc1 = __builtin_amdgcn_wmma_f32_16x16x32_f16(false, a, false, pb1[kt], (short)0, acc1, false, false);
            }
            store_tile(Y, mode, Np, ct, pt0, pt1, m_off, cn, acc0, acc1);
        }
    } else {
        for (int ct = 0; ct < (Cout >> 4); ct += 2) {
            v8f a00, a01, a10, a11;
#pragma unroll
            for (int r = 0; r < 8; ++r) {
                float b0v = bias[ct * 16 + m_off + r];
                float b1v = bias[(ct + 1) * 16 + m_off + r];
                a00[r] = b0v; a01[r] = b0v; a10[r] = b1v; a11[r] = b1v;
            }
#pragma unroll
            for (int kt = 0; kt < KT; ++kt) {
                v16h af0 = load_afrag(W + (long)(ct * 16 + mrow) * CIN + kt * 32, kb);
                v16h af1 = load_afrag(W + (long)((ct + 1) * 16 + mrow) * CIN + kt * 32, kb);
                v16h b0, b1;
#pragma unroll
                for (int s = 0; s < 16; ++s) {
                    const _Float16* xrow = X + (long)(kt * 32 + kb + (s & 7) + ((s >> 3) << 4)) * Np;
                    b0[s] = xrow[pt0 + cn];
                    b1[s] = xrow[pt1 + cn];
                }
                a00 = __builtin_amdgcn_wmma_f32_16x16x32_f16(false, af0, false, b0, (short)0, a00, false, false);
                a01 = __builtin_amdgcn_wmma_f32_16x16x32_f16(false, af0, false, b1, (short)0, a01, false, false);
                a10 = __builtin_amdgcn_wmma_f32_16x16x32_f16(false, af1, false, b0, (short)0, a10, false, false);
                a11 = __builtin_amdgcn_wmma_f32_16x16x32_f16(false, af1, false, b1, (short)0, a11, false, false);
            }
            store_tile(Y, mode, Np, ct,     pt0, pt1, m_off, cn, a00, a01);
            store_tile(Y, mode, Np, ct + 1, pt0, pt1, m_off, cn, a10, a11);
        }
    }
}

// ---------------------------------------------------------------------------
// Fused attention pool (Cc = 128): scores = softmax_k(fcW*fs + fcB),
// agg[c][n] = sum_k fs[c][n,k] * scores[c][n,k].  One wave = 2 points.
// Epilogue: dump the two 16x16 score tiles to a per-wave LDS slot (padded
// rows), then each lane owns one (channel, point): 4x ds_load_b128 of its
// k-row + one 32B fs row load, serial max/exp/weighted-sum in registers.
// Replaces the serialized ds_bpermute shuffle trees entirely.
// ---------------------------------------------------------------------------
#define ROWP 20   // padded row stride (floats): 16B-aligned rows, conflict-free writes

__global__ void att_pool_wmma(const _Float16* __restrict__ fcW, const float* __restrict__ fcB,
                              const _Float16* __restrict__ fs, _Float16* __restrict__ agg) {
    __shared__ float sm[8 * 2 * 16 * ROWP];          // 20 KB, per-wave slots
    const int lane  = threadIdx.x & 31;
    const int wave  = threadIdx.x >> 5;
    const long n0   = ((long)blockIdx.x * 8 + wave) * 2;
    const long pt0  = n0 * 16, pt1 = pt0 + 16;
    const int mrow  = lane & 15;
    const int m_off = (lane < 16) ? 0 : 8;
    const int kb    = (lane < 16) ? 0 : 8;
    const int k     = lane & 15;
    const int t_own = lane >> 4;                     // tile owned for reduction
    const int m_own = lane & 15;                     // channel-in-tile owned
    const long n_own = n0 + t_own;
    float* smw = sm + wave * (2 * 16 * ROWP);
    const float* srow = smw + (t_own * 16 + m_own) * ROWP;

    v16h pb0[4], pb1[4];
#pragma unroll
    for (int kt = 0; kt < 4; ++kt) {
#pragma unroll
        for (int s = 0; s < 16; ++s) {
            const _Float16* xrow = fs + (long)(kt * 32 + kb + (s & 7) + ((s >> 3) << 4)) * NKF;
            pb0[kt][s] = xrow[pt0 + k];
            pb1[kt][s] = xrow[pt1 + k];
        }
    }

    for (int ct = 0; ct < 8; ++ct) {                 // 128 channels / 16
        v8f acc0, acc1;
#pragma unroll
        for (int r = 0; r < 8; ++r) { float bv = fcB[ct * 16 + m_off + r]; acc0[r] = bv; acc1[r] = bv; }
#pragma unroll
        for (int kt = 0; kt < 4; ++kt) {
            v16h a = load_afrag(fcW + (long)(ct * 16 + mrow) * 128 + kt * 32, kb);
            acc0 = __builtin_amdgcn_wmma_f32_16x16x32_f16(false, a, false, pb0[kt], (short)0, acc0, false, false);
            acc1 = __builtin_amdgcn_wmma_f32_16x16x32_f16(false, a, false, pb1[kt], (short)0, acc1, false, false);
        }
        // transpose via LDS (intra-wave: LDS is in-order, no workgroup barrier)
        __builtin_amdgcn_wave_barrier();
#pragma unroll
        for (int r = 0; r < 8; ++r) {
            smw[(m_off + r) * ROWP + k]        = acc0[r];
            smw[(16 + m_off + r) * ROWP + k]   = acc1[r];
        }
        __builtin_amdgcn_wave_barrier();
        const int ch = ct * 16 + m_own;
        v16h fsrow = *(const v16h*)(fs + (long)ch * NKF + n_own * 16);
        float vv[16];
#pragma unroll
        for (int j = 0; j < 4; ++j) {
            float4 q = *(const float4*)(srow + j * 4);
            vv[j * 4 + 0] = q.x; vv[j * 4 + 1] = q.y; vv[j * 4 + 2] = q.z; vv[j * 4 + 3] = q.w;
        }
        __builtin_amdgcn_wave_barrier();
        float mx = vv[0];
#pragma unroll
        for (int kk = 1; kk < 16; ++kk) mx = fmaxf(mx, vv[kk]);
        float se = 0.f, p = 0.f;
#pragma unroll
        for (int kk = 0; kk < 16; ++kk) {
            float e = __expf(vv[kk] - mx);
            se += e;
            p += e * (float)fsrow[kk];
        }
        agg[(long)ch * NPTS + n_own] = (_Float16)(p / se);
    }
}

// ---------------------------------------------------------------------------
extern "C" void kernel_launch(void* const* d_in, const int* in_sizes, int n_in,
                              void* d_out, int out_size, void* d_ws, size_t ws_size,
                              hipStream_t stream) {
    (void)in_sizes; (void)n_in; (void)out_size; (void)ws_size;
    // Inputs in setup_inputs() insertion order; params flattened in nested insertion order.
    const float* xyz   = (const float*)d_in[0];   // [3][16384]
    const float* feat  = (const float*)d_in[1];   // [64][16384]
    const int*   nidx  = (const int*)d_in[2];     // [16384*16]
    auto F = [&](int i) { return (const float*)d_in[i]; };

    size_t off = 0;
    auto alloc = [&](size_t bytes) -> char* {
        off = (off + 255) & ~(size_t)255;
        char* p = (char*)d_ws + off;
        off += bytes;
        return p;
    };
    auto ah = [&](size_t elems) { return (_Float16*)alloc(elems * 2); };
    auto af = [&](size_t elems) { return (float*)alloc(elems * 4); };

    _Float16 *W_pe1 = ah(64*64),   *W_ls1 = ah(64*32),   *W_ls2 = ah(64*64);
    _Float16 *W_fc1 = ah(128*128), *W_a1m1 = ah(64*128), *W_a1m2 = ah(64*64);
    _Float16 *W_pe2a = ah(64*64),  *W_pe2b = ah(64*64);
    _Float16 *W_fc2 = ah(128*128), *W_a2m1 = ah(128*128), *W_a2m2 = ah(128*128);
    _Float16 *W_mo1 = ah(256*128), *W_mo2 = ah(256*256);
    _Float16 *W_sk1 = ah(256*64),  *W_sk2 = ah(256*256);
    float *B_pe1 = af(64), *B_ls1 = af(64), *B_ls2 = af(64), *B_fc1 = af(128);
    float *B_a1m1 = af(64), *B_a1m2 = af(64), *B_pe2a = af(64), *B_pe2b = af(64);
    float *B_fc2 = af(128), *B_a2m1 = af(128), *B_a2m2 = af(128);
    float *B_mo1 = af(256), *B_mo2 = af(256), *B_sk1 = af(256), *B_sk2 = af(256);

    _Float16* ffeat  = ah(64 * NPTS);
    _Float16* pf0    = ah(64 * NPTS);
    _Float16* relpos = ah(32 * NKF);
    _Float16* bufA   = ah(64 * NKF);
    _Float16* fs     = ah(128 * NKF);   // rows 0-63: f_xyz, rows 64-127: gathered pf
    _Float16* agg    = ah(128 * NPTS);
    _Float16* hid_s  = ah(128 * NPTS);
    _Float16* pf1    = ah(64 * NPTS);
    _Float16* pf2    = ah(128 * NPTS);
    _Float16* hid256 = ah(256 * NPTS);

    auto fold = [&](const float* W, const float* b, const float* g, const float* be,
                    _Float16* Wf, float* bf, int Co, int CiS, int CiD) {
        int t = Co * CiD;
        fold_weights<<<(t + 255) / 256, 256, 0, stream>>>(W, b, g, be, Wf, bf, Co, CiS, CiD);
    };
    fold(F(3),  F(4),  F(5),  F(6),  W_pe1, B_pe1, 64, 64, 64);       // pe1 (+BN)
    fold(F(7),  F(8),  nullptr, nullptr, W_ls1, B_ls1, 64, 10, 32);   // locse l1 (pad)
    fold(F(9),  F(10), F(11), F(12), W_ls2, B_ls2, 64, 64, 64);       // locse l2 (+BN)
    fold(F(13), F(14), nullptr, nullptr, W_fc1, B_fc1, 128, 128, 128);
    fold(F(15), F(16), nullptr, nullptr, W_a1m1, B_a1m1, 64, 128, 128);
    fold(F(17), F(18), F(19), F(20), W_a1m2, B_a1m2, 64, 64, 64);
    fold(F(21), F(22), nullptr, nullptr, W_pe2a, B_pe2a, 64, 64, 64);
    fold(F(23), F(24), F(25), F(26), W_pe2b, B_pe2b, 64, 64, 64);
    fold(F(27), F(28), nullptr, nullptr, W_fc2, B_fc2, 128, 128, 128);
    fold(F(29), F(30), nullptr, nullptr, W_a2m1, B_a2m1, 128, 128, 128);
    fold(F(31), F(32), F(33), F(34), W_a2m2, B_a2m2, 128, 128, 128);
    fold(F(35), F(36), nullptr, nullptr, W_mo1, B_mo1, 256, 128, 128);
    fold(F(37), F(38), F(39), F(40), W_mo2, B_mo2, 256, 256, 256);
    fold(F(41), F(42), nullptr, nullptr, W_sk1, B_sk1, 256, 64, 64);
    fold(F(43), F(44), F(45), F(46), W_sk2, B_sk2, 256, 256, 256);

    f32_to_f16<<<(int)((64 * NPTS + 255) / 256), 256, 0, stream>>>(feat, ffeat, 64 * NPTS);

    auto conv = [&](const _Float16* W, const float* b, const _Float16* X, void* Y,
                    int Co, int Ci, long Np, int mode) {
        int grid = (int)(Np / 256);
        switch (Ci) {
            case 32:  conv1x1_wmma<32> <<<grid, 256, 0, stream>>>(W, b, X, Y, Co, Np, mode); break;
            case 64:  conv1x1_wmma<64> <<<grid, 256, 0, stream>>>(W, b, X, Y, Co, Np, mode); break;
            case 128: conv1x1_wmma<128><<<grid, 256, 0, stream>>>(W, b, X, Y, Co, Np, mode); break;
            default:  conv1x1_wmma<256><<<grid, 256, 0, stream>>>(W, b, X, Y, Co, Np, mode); break;
        }
    };

    // --- pipeline ---
    conv(W_pe1, B_pe1, ffeat, pf0, 64, 64, NPTS, 0);                       // pe1
    build_relpos<<<(int)(NKF / 256), 256, 0, stream>>>(xyz, nidx, relpos); // LocSE relpos
    conv(W_ls1, B_ls1, relpos, bufA, 64, 32, NKF, 0);                      // locse l1
    conv(W_ls2, B_ls2, bufA, fs, 64, 64, NKF, 0);                          // f_xyz -> fs[0:64]
    gather_rows<<<(int)(NKF / 256), 256, 0, stream>>>(pf0, nidx, fs + 64 * NKF, 64);
    att_pool_wmma<<<(int)(NPTS / 16), 256, 0, stream>>>(W_fc1, B_fc1, fs, agg);
    conv(W_a1m1, B_a1m1, agg, hid_s, 64, 128, NPTS, 0);                    // ap1 mlp
    conv(W_a1m2, B_a1m2, hid_s, pf1, 64, 64, NPTS, 0);
    conv(W_pe2a, B_pe2a, fs, bufA, 64, 64, NKF, 0);                        // pe2 on f_xyz
    conv(W_pe2b, B_pe2b, bufA, fs, 64, 64, NKF, 0);                        // -> fs[0:64]
    gather_rows<<<(int)(NKF / 256), 256, 0, stream>>>(pf1, nidx, fs + 64 * NKF, 64);
    att_pool_wmma<<<(int)(NPTS / 16), 256, 0, stream>>>(W_fc2, B_fc2, fs, agg);
    conv(W_a2m1, B_a2m1, agg, hid_s, 128, 128, NPTS, 0);                   // ap2 mlp
    conv(W_a2m2, B_a2m2, hid_s, pf2, 128, 128, NPTS, 0);
    conv(W_mo1, B_mo1, pf2, hid256, 256, 128, NPTS, 0);                    // mlp_out
    conv(W_mo2, B_mo2, hid256, d_out, 256, 256, NPTS, 1);                  // f32, no act
    conv(W_sk1, B_sk1, ffeat, hid256, 256, 64, NPTS, 0);                   // skip
    conv(W_sk2, B_sk2, hid256, d_out, 256, 256, NPTS, 2);                  // add + lrelu
}
